// MultiHeadAttention_11605001634068
// MI455X (gfx1250) — matrix operand order repaired and checked by reading, not compiled, not run on hardware
//
#include <hip/hip_runtime.h>

typedef __attribute__((ext_vector_type(16))) _Float16 v16h;
typedef __attribute__((ext_vector_type(8)))  _Float16 v8h;
typedef __attribute__((ext_vector_type(8)))  float    v8f;
typedef __attribute__((ext_vector_type(4)))  float    v4f;

#define B_  2
#define S_  2048
#define D_  512
#define H_  8
#define K_  64

static __device__ inline v8f wmma16(v16h a, v16h b, v8f c) {
  // D = A(16x32 f16) * B(32x16 f16) + C(16x16 f32)
  return __builtin_amdgcn_wmma_f32_16x16x32_f16(false, a, false, b, (short)0, c,
                                                false, false);
}

static __device__ inline float redmax16(float x) {
#pragma unroll
  for (int m = 1; m < 16; m <<= 1) x = fmaxf(x, __shfl_xor(x, m, 32));
  return x;
}
static __device__ inline float redsum16(float x) {
#pragma unroll
  for (int m = 1; m < 16; m <<= 1) x += __shfl_xor(x, m, 32);
  return x;
}

// ---------------------------------------------------------------------------
// Kernel 1: per-head QKV projections.
//   z=0: Qh[bh][s][k] = (query @ Wq[h] + bq[h]) * 1/sqrt(K)   (f16)
//   z=1: Kh[bh][s][k] =  key   @ Wk[h] + bk[h]                (f16)
//   z=2: Vt[bh][k][s] = (value @ Wv[h] + bv[h])^T             (f16, transposed)
// Block: 256 threads = 8 waves; each wave computes a 16x64 tile.
// ---------------------------------------------------------------------------
__global__ __launch_bounds__(256) void proj_qkv(
    const float* __restrict__ q, const float* __restrict__ k,
    const float* __restrict__ v, const float* __restrict__ Wq,
    const float* __restrict__ Wk, const float* __restrict__ Wv,
    const float* __restrict__ bq, const float* __restrict__ bk,
    const float* __restrict__ bv, _Float16* __restrict__ Qh,
    _Float16* __restrict__ Kh, _Float16* __restrict__ Vt) {
  __shared__ __align__(64) _Float16 Wt[64 * 32];  // [n][dloc] (transposed)

  const int z = blockIdx.z;
  const int bh = blockIdx.y;
  const int b = bh >> 3, h = bh & 7;
  const int tid = threadIdx.x;
  const int lane = tid & 31, w = tid >> 5;
  const int ln = lane & 15, hi = lane >> 4;

  const float* X = (z == 0) ? q : (z == 1) ? k : v;
  const float* W = (z == 0) ? Wq : (z == 1) ? Wk : Wv;
  const float* bias = (z == 0) ? bq : (z == 1) ? bk : bv;

  const int s0 = blockIdx.x * 128;
  const int rowA = s0 + w * 16 + ln;
  const float* Arow = X + ((size_t)b * S_ + rowA) * D_;
  const float* Wbase = W + (size_t)h * D_ * K_;

  v8f acc[4] = {{}, {}, {}, {}};

  for (int c = 0; c < D_ / 32; ++c) {
    __syncthreads();
    // cooperative stage of the 32x64 weight chunk, transposed + f16
#pragma unroll
    for (int j = 0; j < 8; ++j) {
      int lin = tid + 256 * j;
      int dloc = lin >> 6, n = lin & 63;
      Wt[n * 32 + dloc] = (_Float16)Wbase[(size_t)(c * 32 + dloc) * K_ + n];
    }
    __syncthreads();
    // A fragment (16x32, contiguous k = 16*hi + e) straight from global f32 -> f16
    const float* ap = Arow + c * 32 + 16 * hi;
    v4f f0 = *(const v4f*)(ap);
    v4f f1 = *(const v4f*)(ap + 4);
    v4f f2 = *(const v4f*)(ap + 8);
    v4f f3 = *(const v4f*)(ap + 12);
    v16h a;
#pragma unroll
    for (int e = 0; e < 4; ++e) {
      a[e] = (_Float16)f0[e];
      a[e + 4] = (_Float16)f1[e];
      a[e + 8] = (_Float16)f2[e];
      a[e + 12] = (_Float16)f3[e];
    }
#pragma unroll
    for (int f = 0; f < 4; ++f) {
      const v16h bfrag = *(const v16h*)&Wt[(16 * f + ln) * 32 + 16 * hi];
      acc[f] = wmma16(a, bfrag, acc[f]);
    }
  }

  const float scale = (z == 0) ? 0.125f : 1.0f;  // 1/sqrt(64) folded into Q
#pragma unroll
  for (int f = 0; f < 4; ++f) {
    const int n = 16 * f + ln;
    const float bb = bias[h * K_ + n];
#pragma unroll
    for (int r = 0; r < 8; ++r) {
      const int row = s0 + w * 16 + r + 8 * hi;
      const float val = (acc[f][r] + bb) * scale;
      if (z == 0)
        Qh[((size_t)bh * S_ + row) * K_ + n] = (_Float16)val;
      else if (z == 1)
        Kh[((size_t)bh * S_ + row) * K_ + n] = (_Float16)val;
      else
        Vt[((size_t)bh * K_ + n) * S_ + row] = (_Float16)val;
    }
  }
}

// ---------------------------------------------------------------------------
// Kernel 2: flash-style attention. Each wave owns 16 query rows, streams all
// 2048 keys in chunks of 32 with online softmax. ctx written f16 as
// [b][s][h*K+k] so the output projection reads it as a plain [B*S,512] GEMM.
// ---------------------------------------------------------------------------
__global__ __launch_bounds__(256) void attn(
    const _Float16* __restrict__ Qh, const _Float16* __restrict__ Kh,
    const _Float16* __restrict__ Vt, _Float16* __restrict__ ctx) {
  __shared__ __align__(64) _Float16 Pl[8 * 16 * 32];  // per-wave P transpose tile

  const int bh = blockIdx.y;
  const int b = bh >> 3, h = bh & 7;
  const int tid = threadIdx.x, lane = tid & 31, w = tid >> 5;
  const int ln = lane & 15, hi = lane >> 4;
  const int q0 = blockIdx.x * 128 + w * 16;

  // Q (16x64) held in registers as two A-fragments
  const _Float16* qrow = Qh + ((size_t)bh * S_ + q0 + ln) * K_;
  v16h qf[2];
#pragma unroll
  for (int c = 0; c < 2; ++c) {
    const v8h lo = *(const v8h*)(qrow + c * 32 + 16 * hi);
    const v8h h8 = *(const v8h*)(qrow + c * 32 + 8 + 16 * hi);
#pragma unroll
    for (int e = 0; e < 8; ++e) {
      qf[c][e] = lo[e];
      qf[c][e + 8] = h8[e];
    }
  }

  float m[8], l[8];
#pragma unroll
  for (int r = 0; r < 8; ++r) {
    m[r] = -1e30f;
    l[r] = 0.f;
  }
  v8f acc[4] = {{}, {}, {}, {}};

  _Float16* pw = &Pl[w * 512];

  for (int t0 = 0; t0 < S_; t0 += 32) {
    if (t0 + 32 < S_) {  // uniform branch: prefetch next K/V chunk
      __builtin_prefetch(Kh + ((size_t)bh * S_ + t0 + 32 + lane) * K_, 0, 1);
      __builtin_prefetch(Vt + ((size_t)bh * K_ + lane) * S_ + t0 + 32, 0, 1);
    }
    // scores S = Q * K^T for 32 keys (K^T B-fragments == Kh row-major reads)
    v8f s[2];
#pragma unroll
    for (int nt = 0; nt < 2; ++nt) {
      s[nt] = (v8f){};
#pragma unroll
      for (int c = 0; c < 2; ++c) {
        const v16h kf = *(const v16h*)(Kh +
            ((size_t)bh * S_ + t0 + 16 * nt + ln) * K_ + c * 32 + 16 * hi);
        s[nt] = wmma16(qf[c], kf, s[nt]);
      }
    }
    // online softmax
    float corr[8];
#pragma unroll
    for (int r = 0; r < 8; ++r) {
      float mx = redmax16(fmaxf(s[0][r], s[1][r]));
      const float nm = fmaxf(m[r], mx);
      const float p0 = __expf(s[0][r] - nm);
      const float p1 = __expf(s[1][r] - nm);
      pw[(r + 8 * hi) * 32 + ln] = (_Float16)p0;
      pw[(r + 8 * hi) * 32 + 16 + ln] = (_Float16)p1;
      const float rs = redsum16(p0 + p1);
      const float cf = __expf(m[r] - nm);
      l[r] = l[r] * cf + rs;
      m[r] = nm;
      corr[r] = cf;
    }
#pragma unroll
    for (int f = 0; f < 4; ++f)
#pragma unroll
      for (int r = 0; r < 8; ++r) acc[f][r] *= corr[r];
    __syncthreads();
    // re-load P in A-fragment layout
    v16h pf;
    {
      const v8h lo = *(const v8h*)(pw + ln * 32 + 16 * hi);
      const v8h h8 = *(const v8h*)(pw + ln * 32 + 8 + 16 * hi);
#pragma unroll
      for (int e = 0; e < 8; ++e) {
        pf[e] = lo[e];
        pf[e + 8] = h8[e];
      }
    }
    // acc += P(16x32) * V(32x64); Vt rows give contiguous-t B-fragments
#pragma unroll
    for (int f = 0; f < 4; ++f) {
      const v16h vf = *(const v16h*)(Vt +
          ((size_t)bh * K_ + 16 * f + ln) * S_ + t0 + 16 * hi);
      acc[f] = wmma16(pf, vf, acc[f]);
    }
    __syncthreads();
  }
  // normalize and store ctx (f16) as [b][s][h*K+k]
#pragma unroll
  for (int r = 0; r < 8; ++r) {
    const float inv = 1.0f / l[r];
    const int row = q0 + r + 8 * hi;
#pragma unroll
    for (int f = 0; f < 4; ++f) {
      const int n = h * K_ + 16 * f + ln;
      ctx[((size_t)b * S_ + row) * (H_ * K_) + n] = (_Float16)(acc[f][r] * inv);
    }
  }
}

// ---------------------------------------------------------------------------
// Kernel 3: output projection  out[B*S,512] = ctx(f16) @ Wo + bo   (f32 out)
// ---------------------------------------------------------------------------
__global__ __launch_bounds__(256) void out_proj(
    const _Float16* __restrict__ ctx, const float* __restrict__ Wo,
    const float* __restrict__ bo, float* __restrict__ out) {
  __shared__ __align__(64) _Float16 Wt[64 * 32];

  const int tid = threadIdx.x, lane = tid & 31, w = tid >> 5;
  const int ln = lane & 15, hi = lane >> 4;
  const int r0 = blockIdx.x * 128;
  const int n0 = blockIdx.y * 64;
  const int rowA = r0 + w * 16 + ln;
  const _Float16* Arow = ctx + (size_t)rowA * (H_ * K_);

  v8f acc[4] = {{}, {}, {}, {}};

  for (int c = 0; c < (H_ * K_) / 32; ++c) {
    __syncthreads();
#pragma unroll
    for (int j = 0; j < 8; ++j) {
      int lin = tid + 256 * j;
      int dloc = lin >> 6, n = lin & 63;
      Wt[n * 32 + dloc] = (_Float16)Wo[(size_t)(c * 32 + dloc) * D_ + n0 + n];
    }
    __syncthreads();
    v16h a;
    {
      const v8h lo = *(const v8h*)(Arow + c * 32 + 16 * hi);
      const v8h h8 = *(const v8h*)(Arow + c * 32 + 8 + 16 * hi);
#pragma unroll
      for (int e = 0; e < 8; ++e) {
        a[e] = lo[e];
        a[e + 8] = h8[e];
      }
    }
#pragma unroll
    for (int f = 0; f < 4; ++f) {
      const v16h bfrag = *(const v16h*)&Wt[(16 * f + ln) * 32 + 16 * hi];
      acc[f] = wmma16(a, bfrag, acc[f]);
    }
  }
#pragma unroll
  for (int f = 0; f < 4; ++f) {
    const int n = n0 + 16 * f + ln;
    const float bias = bo[n];
#pragma unroll
    for (int r = 0; r < 8; ++r) {
      const int row = r0 + w * 16 + r + 8 * hi;
      out[(size_t)row * D_ + n] = acc[f][r] + bias;
    }
  }
}

extern "C" void kernel_launch(void* const* d_in, const int* in_sizes, int n_in,
                              void* d_out, int out_size, void* d_ws,
                              size_t ws_size, hipStream_t stream) {
  const float* q = (const float*)d_in[0];
  const float* k = (const float*)d_in[1];
  const float* v = (const float*)d_in[2];
  const float* Wq = (const float*)d_in[3];
  const float* bq = (const float*)d_in[4];
  const float* Wk = (const float*)d_in[5];
  const float* bk = (const float*)d_in[6];
  const float* Wv = (const float*)d_in[7];
  const float* bv = (const float*)d_in[8];
  const float* Wo = (const float*)d_in[9];
  const float* bo = (const float*)d_in[10];
  float* out = (float*)d_out;

  const size_t nBHSK = (size_t)B_ * H_ * S_ * K_;  // 2M f16 elems = 4 MB each
  _Float16* Qh = (_Float16*)d_ws;
  _Float16* Kh = Qh + nBHSK;
  _Float16* Vt = Kh + nBHSK;
  _Float16* ctx = Vt + nBHSK;  // total ws use: 16 MB

  proj_qkv<<<dim3(S_ / 128, B_ * H_, 3), 256, 0, stream>>>(
      q, k, v, Wq, Wk, Wv, bq, bk, bv, Qh, Kh, Vt);
  attn<<<dim3(S_ / 128, B_ * H_), 256, 0, stream>>>(Qh, Kh, Vt, ctx);
  out_proj<<<dim3((B_ * S_) / 128, D_ / 64), 256, 0, stream>>>(ctx, Wo, bo, out);
}
